// RestaurantGNN_11596411699471
// MI455X (gfx1250) — matrix operand order repaired and checked by reading, not compiled, not run on hardware
//
#include <hip/hip_runtime.h>

// ---------------------------------------------------------------------------
// CDNA5 (gfx1250) implementation: AGNNConv + TransformerEncoderLayer + head.
// Dense matmuls: v_wmma_f32_16x16x32_bf16 (wave32 WMMA).
// Tile staging: global_load_async_to_lds_b128 (ASYNCcnt), ping-pong double
// buffered so copies for tile s+1 overlap WMMA on tile s.
// ---------------------------------------------------------------------------

typedef __attribute__((ext_vector_type(16))) __bf16 v16bf;
typedef __attribute__((ext_vector_type(8)))  float  v8f;

union Frag16 { unsigned u[8]; v16bf v; };

__device__ inline unsigned short f2bf(float f) {
  unsigned u = __float_as_uint(f);
  unsigned r = u + 0x7FFFu + ((u >> 16) & 1u);   // round-to-nearest-even
  return (unsigned short)(r >> 16);
}

__device__ inline v8f wmma_bf16(v16bf a, v16bf b, v8f c) {
  return __builtin_amdgcn_wmma_f32_16x16x32_bf16(
      false, a, false, b, (short)0, c, false, false);
}

// Low 32 bits of a generic pointer to __shared__ data == LDS byte offset
// (ISA 10.2: LDS aperture address truncates to addr[31:0]).
__device__ inline unsigned lds_off(const void* p) {
  return (unsigned)(unsigned long long)p;
}

// Async copy 16B global -> LDS, GVS addressing (saddr + 32b voffset).
__device__ inline void async_b128(unsigned lds, unsigned voff, const void* sbase) {
  asm volatile("global_load_async_to_lds_b128 %0, %1, %2"
               :: "v"(lds), "v"(voff), "s"(sbase) : "memory");
}
__device__ inline void wait_async0() { asm volatile("s_wait_asynccnt 0x0" ::: "memory"); }
__device__ inline void wait_async2() { asm volatile("s_wait_asynccnt 0x2" ::: "memory"); }
__device__ inline void wait_async4() { asm volatile("s_wait_asynccnt 0x4" ::: "memory"); }

// A-matrix 16x32 bf16 fragment gather from LDS tile, row-major [16][ldk].
// lane m=l%16; VGPR v holds K = (v/4)*16 + (l/16)*8 + (v%4)*2 (+1).
__device__ inline v16bf ldsGatherA(const unsigned short* t, int lane, int ldk, int koff) {
  Frag16 f;
  int m = lane & 15, hi = lane >> 4;
#pragma unroll
  for (int v = 0; v < 8; ++v) {
    int kb = ((v >> 2) << 4) + (hi << 3) + ((v & 3) << 1);
    f.u[v] = *(const unsigned*)(t + m * ldk + koff + kb);
  }
  return f.v;
}

// B-matrix 32x16 bf16 fragment gather from LDS tile stored as B^T row-major
// [16 n-rows][ldk k-cols]. lane n=l%16; VGPR v holds K = (v/4)*16 + (l/16)*8 + (v%4)*2 (+1).
__device__ inline v16bf ldsGatherB(const unsigned short* t, int lane, int ldk, int koff) {
  Frag16 f;
  int n = lane & 15, hi = lane >> 4;
#pragma unroll
  for (int v = 0; v < 8; ++v) {
    int kb = ((v >> 2) << 4) + (hi << 3) + ((v & 3) << 1);
    f.u[v] = *(const unsigned*)(t + n * ldk + koff + kb);
  }
  return f.v;
}

// fp32 -> bf16 conversion pass (activations / weights)
__global__ __launch_bounds__(256) void cvt_bf16(const float* __restrict__ s,
                                                unsigned short* __restrict__ d,
                                                int n) {
  int i = blockIdx.x * 256 + threadIdx.x;
  if (i < n) d[i] = f2bf(s[i]);
}

// ---------------------------------------------------------------------------
// GEMM: C[M,N] = A[M,K] @ W[N,K]^T + bias[N]; A,W bf16; C f32 or bf16.
// Block = 128 threads (4 waves); block tile 64x64; wave owns 16x64 (4 WMMA
// per A fragment). A/W tiles async-staged, ping-pong double buffered.
// ---------------------------------------------------------------------------
template <int RELU, int OUTBF>
__global__ __launch_bounds__(128) void gemm_bf16(
    const unsigned short* __restrict__ A, const unsigned short* __restrict__ W,
    const float* __restrict__ bias, void* __restrict__ Cout,
    int M, int N, int K) {
  __shared__ __align__(16) unsigned short sA[2][64 * 32];
  __shared__ __align__(16) unsigned short sB[2][64 * 32];
  int t = threadIdx.x, lane = t & 31, wv = t >> 5;
  int m0 = blockIdx.x * 64, n0 = blockIdx.y * 64;
  bool fullW = (n0 + 64 <= N);          // uniform per block
  v8f vz = {};
  v8f acc[4];
#pragma unroll
  for (int f = 0; f < 4; ++f) acc[f] = vz;
  unsigned sAoff = lds_off(&sA[0][0]);
  unsigned sBoff = lds_off(&sB[0][0]);
  const int nsteps = K >> 5;

  auto stage = [&](int k0, int buf) {
    const char* abase = (const char*)A + ((size_t)m0 * K + k0) * 2;
#pragma unroll
    for (int i = 0; i < 2; ++i) {       // 64 rows x 64B = 4KB async
      int lb = (t + i * 128) * 16;
      int r = lb >> 6, cb = lb & 63;
      async_b128(sAoff + buf * 4096 + lb, (unsigned)(r * K * 2 + cb), abase);
    }
    if (fullW) {
      const char* wbase = (const char*)W + ((size_t)n0 * K + k0) * 2;
#pragma unroll
      for (int i = 0; i < 2; ++i) {
        int lb = (t + i * 128) * 16;
        int r = lb >> 6, cb = lb & 63;
        async_b128(sBoff + buf * 4096 + lb, (unsigned)(r * K * 2 + cb), wbase);
      }
    } else {                            // ragged N (head): guarded copy
#pragma unroll
      for (int i = 0; i < 16; ++i) {
        int idx = t * 16 + i;
        int r = idx >> 5, c = idx & 31;
        sB[buf][idx] = (n0 + r < N) ? W[(size_t)(n0 + r) * K + k0 + c]
                                    : (unsigned short)0;
      }
    }
  };

  stage(0, 0);
  for (int s = 0; s < nsteps; ++s) {
    int cur = s & 1;
    if (s + 1 < nsteps) {
      stage((s + 1) * 32, cur ^ 1);     // stream next tile behind compute
      if (fullW) wait_async4(); else wait_async2();
    } else {
      wait_async0();
    }
    __syncthreads();
    v16bf a = ldsGatherA(sA[cur] + wv * 16 * 32, lane, 32, 0);
#pragma unroll
    for (int f = 0; f < 4; ++f) {
      v16bf b = ldsGatherB(sB[cur] + f * 16 * 32, lane, 32, 0);
      acc[f] = wmma_bf16(a, b, acc[f]);
    }
    __syncthreads();                    // protect cur before next overwrite
  }

  int n = lane & 15, hi = lane >> 4;
#pragma unroll
  for (int f = 0; f < 4; ++f) {
    int col = n0 + f * 16 + n;
    float bv = (col < N) ? bias[col] : 0.0f;
#pragma unroll
    for (int v = 0; v < 8; ++v) {
      int m = m0 + wv * 16 + v + hi * 8;
      float val = acc[f][v] + bv;
      if (RELU) val = fmaxf(val, 0.0f);
      if (col < N && m < M) {
        if (OUTBF) ((unsigned short*)Cout)[(size_t)m * N + col] = f2bf(val);
        else       ((float*)Cout)[(size_t)m * N + col] = val;
      }
    }
  }
}

// ---------------------------------------------------------------------------
// Flash attention: N=8192, 2 heads, dh=128; qkv packed bf16 [N][768].
// Block = 128 threads (4 waves); wave owns 16 queries; 32-key chunks,
// K async-staged + V transposed, both ping-pong double buffered.
// ---------------------------------------------------------------------------
__global__ __launch_bounds__(128) void attn_flash(
    const unsigned short* __restrict__ qkvb, unsigned short* __restrict__ outp) {
  const int NN = 8192, C3 = 768, DH = 128;
  __shared__ __align__(16) unsigned short sK[2][32 * 128];   // [key][dim]
  __shared__ __align__(16) unsigned short sV[2][128 * 32];   // [dim][key]
  __shared__ __align__(16) unsigned short sP[4][16 * 32];    // per-wave P
  int t = threadIdx.x, lane = t & 31, wv = t >> 5;
  int h = blockIdx.y;
  int q0 = blockIdx.x * 64 + wv * 16;
  int qoff = h * DH, koff = 256 + h * DH, voff = 512 + h * DH;
  const float scale = 0.08838834764831845f; // 1/sqrt(128)
  unsigned sKoff = lds_off(&sK[0][0]);

  auto stageK = [&](int kb, int buf) {   // 32 keys x 256B = 8KB async
    const char* kbase = (const char*)qkvb + ((size_t)kb * C3 + koff) * 2;
#pragma unroll
    for (int i = 0; i < 4; ++i) {
      int lb = (t + i * 128) * 16;
      int key = lb >> 8, db = lb & 255;
      async_b128(sKoff + buf * 8192 + lb, (unsigned)(key * (C3 * 2) + db), kbase);
    }
  };
  auto stageV = [&](int kb, int buf) {   // transpose: sV[dim][key]
#pragma unroll 4
    for (int i = 0; i < 32; ++i) {
      int idx = t + i * 128;
      int key = idx >> 7, d = idx & 127;
      sV[buf][d * 32 + key] = qkvb[(size_t)(kb + key) * C3 + voff + d];
    }
  };

  // Resident Q fragments (bf16 in global -> direct aligned b32 gathers)
  v16bf qf[4];
  {
    int m = lane & 15, hi = lane >> 4;
    const unsigned short* qrow = qkvb + (size_t)(q0 + m) * C3 + qoff;
#pragma unroll
    for (int c = 0; c < 4; ++c) {
      Frag16 f;
#pragma unroll
      for (int v = 0; v < 8; ++v) {
        int kb = c * 32 + ((v >> 2) << 4) + (hi << 3) + ((v & 3) << 1);
        f.u[v] = *(const unsigned*)(qrow + kb);
      }
      qf[c] = f.v;
    }
  }

  float mstate[8], lstate[8];
  v8f acc[8];
  v8f vz = {};
#pragma unroll
  for (int i = 0; i < 8; ++i) { mstate[i] = -1e30f; lstate[i] = 0.0f; acc[i] = vz; }

  stageK(0, 0);
  stageV(0, 0);
  for (int kb = 0; kb < NN; kb += 32) {
    int cur = (kb >> 5) & 1;
    if (kb + 32 < NN) {
      stageK(kb + 32, cur ^ 1);
      stageV(kb + 32, cur ^ 1);
      wait_async4();                    // cur's 4 copies done; next in flight
    } else {
      wait_async0();
    }
    __syncthreads();

    // S = Q @ K^T for two 16-key subtiles (4 chained WMMA over dh=128)
    v8f S[2];
#pragma unroll
    for (int sub = 0; sub < 2; ++sub) {
      v8f s = vz;
#pragma unroll
      for (int c = 0; c < 4; ++c) {
        v16bf b = ldsGatherB(sK[cur] + sub * 16 * 128, lane, 128, c * 32);
        s = wmma_bf16(qf[c], b, s);
      }
      S[sub] = s;
    }

    // Online softmax (row = v + 8*(lane/16); reduce across 16 lanes)
#pragma unroll
    for (int v = 0; v < 8; ++v) {
      float s0 = S[0][v] * scale, s1 = S[1][v] * scale;
      float mx = fmaxf(s0, s1);
#pragma unroll
      for (int d = 8; d >= 1; d >>= 1) mx = fmaxf(mx, __shfl_xor(mx, d, 32));
      float mnew = fmaxf(mstate[v], mx);
      float sc = __expf(mstate[v] - mnew);
      float p0 = __expf(s0 - mnew);
      float p1 = __expf(s1 - mnew);
      float ps = p0 + p1;
#pragma unroll
      for (int d = 8; d >= 1; d >>= 1) ps += __shfl_xor(ps, d, 32);
      lstate[v] = lstate[v] * sc + ps;
      mstate[v] = mnew;
      S[0][v] = p0;
      S[1][v] = p1;
#pragma unroll
      for (int f = 0; f < 8; ++f) acc[f][v] *= sc;
    }

    // P (C-layout f32) -> LDS 16x32 bf16 -> A fragment; acc += P @ V
    {
      int n = lane & 15, hi = lane >> 4;
      unsigned short* P = sP[wv];
#pragma unroll
      for (int v = 0; v < 8; ++v) {
        int m = v + hi * 8;
        P[m * 32 + n]      = f2bf(S[0][v]);
        P[m * 32 + 16 + n] = f2bf(S[1][v]);
      }
      v16bf pa = ldsGatherA(P, lane, 32, 0);
#pragma unroll
      for (int f = 0; f < 8; ++f) {
        v16bf b = ldsGatherB(sV[cur] + f * 16 * 32, lane, 32, 0);
        acc[f] = wmma_bf16(pa, b, acc[f]);
      }
    }
    __syncthreads();
  }

  // Epilogue: normalize; write bf16 attn output [N][256] (head concat)
  int n = lane & 15, hi = lane >> 4;
#pragma unroll
  for (int v = 0; v < 8; ++v) {
    int m = q0 + v + hi * 8;
    float inv = 1.0f / lstate[v];
#pragma unroll
    for (int f = 0; f < 8; ++f)
      outp[(size_t)m * 256 + h * 128 + f * 16 + n] = f2bf(acc[f][v] * inv);
  }
}

// ---------------------------------------------------------------------------
// AGNN scatter pipeline (f32, bandwidth/atomic bound)
// ---------------------------------------------------------------------------
__global__ __launch_bounds__(256) void l2norm_rows(const float* __restrict__ x,
                                                   float* __restrict__ xn) {
  __shared__ float red[8];
  __shared__ float stot;
  int r = blockIdx.x, t = threadIdx.x;
  float v = x[(size_t)r * 256 + t];
  float ss = v * v;
#pragma unroll
  for (int m = 16; m >= 1; m >>= 1) ss += __shfl_xor(ss, m, 32);
  if ((t & 31) == 0) red[t >> 5] = ss;
  __syncthreads();
  if (t == 0) { float s = 0; for (int i = 0; i < 8; ++i) s += red[i]; stot = s; }
  __syncthreads();
  float norm = sqrtf(stot) + 1e-12f;
  xn[(size_t)r * 256 + t] = v / norm;
}

__global__ __launch_bounds__(256) void edge_alpha(
    const float* __restrict__ xn, const int* __restrict__ ei,
    const float* __restrict__ beta, float* __restrict__ alpha,
    unsigned* __restrict__ mxu, int E, int E2) {
  int e = blockIdx.x * 8 + (threadIdx.x >> 5);
  if (e >= E2) return;
  int lane = threadIdx.x & 31;
  int s, d;
  if (e < E) { s = ei[e]; d = ei[E + e]; } else { s = d = e - E; }
  const float* ps = xn + (size_t)s * 256;
  const float* pd = xn + (size_t)d * 256;
  float dot = 0.0f;
#pragma unroll
  for (int i = 0; i < 8; ++i) dot += ps[lane + i * 32] * pd[lane + i * 32];
#pragma unroll
  for (int m = 16; m >= 1; m >>= 1) dot += __shfl_xor(dot, m, 32);
  if (lane == 0) {
    float a = beta[0] * dot;
    alpha[e] = a;
    unsigned b = __float_as_uint(a);
    unsigned key = (b & 0x80000000u) ? ~b : (b | 0x80000000u);
    atomicMax(mxu + d, key);
  }
}

__global__ __launch_bounds__(256) void edge_exp(
    const float* __restrict__ alpha, const int* __restrict__ ei,
    const unsigned* __restrict__ mxu, float* __restrict__ ex,
    float* __restrict__ den, int E, int E2) {
  int e = blockIdx.x * 256 + threadIdx.x;
  if (e >= E2) return;
  int d = (e < E) ? ei[E + e] : (e - E);
  unsigned u = mxu[d];
  unsigned b = (u & 0x80000000u) ? (u & 0x7FFFFFFFu) : ~u;
  float mx = __uint_as_float(b);
  float v = __expf(alpha[e] - mx);
  ex[e] = v;
  atomicAdd(den + d, v);
}

__global__ __launch_bounds__(256) void edge_scatter(
    const float* __restrict__ x, const int* __restrict__ ei,
    const float* __restrict__ ex, const float* __restrict__ den,
    float* __restrict__ h, int E, int E2) {
  int e = blockIdx.x * 8 + (threadIdx.x >> 5);
  if (e >= E2) return;
  int lane = threadIdx.x & 31;
  int s, d;
  if (e < E) { s = ei[e]; d = ei[E + e]; } else { s = d = e - E; }
  float w = ex[e] / den[d];
#pragma unroll
  for (int i = 0; i < 8; ++i)
    atomicAdd(h + (size_t)d * 256 + lane + i * 32,
              w * x[(size_t)s * 256 + lane + i * 32]);
}

// out = LayerNorm(a + b) * g + bb  (row length 256); writes f32 and/or bf16
__global__ __launch_bounds__(256) void ln_res(
    const float* __restrict__ a, const float* __restrict__ b,
    const float* __restrict__ g, const float* __restrict__ bb,
    float* __restrict__ o, unsigned short* __restrict__ obf) {
  __shared__ float red[8];
  __shared__ float smean, svar;
  int r = blockIdx.x, t = threadIdx.x;
  size_t idx = (size_t)r * 256 + t;
  float v = a[idx] + b[idx];
  float s = v;
#pragma unroll
  for (int m = 16; m >= 1; m >>= 1) s += __shfl_xor(s, m, 32);
  if ((t & 31) == 0) red[t >> 5] = s;
  __syncthreads();
  if (t == 0) { float tot = 0; for (int i = 0; i < 8; ++i) tot += red[i]; smean = tot * (1.0f / 256.0f); }
  __syncthreads();
  float c = v - smean;
  float sq = c * c;
#pragma unroll
  for (int m = 16; m >= 1; m >>= 1) sq += __shfl_xor(sq, m, 32);
  if ((t & 31) == 0) red[t >> 5] = sq;
  __syncthreads();
  if (t == 0) { float tot = 0; for (int i = 0; i < 8; ++i) tot += red[i]; svar = tot * (1.0f / 256.0f); }
  __syncthreads();
  float res = c * rsqrtf(svar + 1e-5f) * g[t] + bb[t];
  if (o)   o[idx] = res;
  if (obf) obf[idx] = f2bf(res);
}

// ---------------------------------------------------------------------------
extern "C" void kernel_launch(void* const* d_in, const int* in_sizes, int n_in,
                              void* d_out, int out_size, void* d_ws, size_t ws_size,
                              hipStream_t stream) {
  const int NN = 8192, D = 256, E = 262144, FF = 2048, HD = 512;
  const int E2 = E + NN;
  const float* x    = (const float*)d_in[0];
  const int*   ei   = (const int*)d_in[1];
  const float* beta = (const float*)d_in[2];
  const float* inW  = (const float*)d_in[3];
  const float* inB  = (const float*)d_in[4];
  const float* opW  = (const float*)d_in[5];
  const float* opB  = (const float*)d_in[6];
  const float* g1   = (const float*)d_in[7];
  const float* b1   = (const float*)d_in[8];
  const float* g2   = (const float*)d_in[9];
  const float* b2   = (const float*)d_in[10];
  const float* f1W  = (const float*)d_in[11];
  const float* f1B  = (const float*)d_in[12];
  const float* f2W  = (const float*)d_in[13];
  const float* f2B  = (const float*)d_in[14];
  const float* fcW  = (const float*)d_in[15];
  const float* fcB  = (const float*)d_in[16];
  const float* oW   = (const float*)d_in[17];
  const float* oB   = (const float*)d_in[18];
  float* out = (float*)d_out;
  (void)in_sizes; (void)n_in; (void)out_size; (void)ws_size;

  char* base = (char*)d_ws;
  size_t off = 0;
  auto carve = [&](size_t bytes) -> void* {
    void* p = base + off;
    off += (bytes + 255) & ~(size_t)255;
    return p;
  };
  typedef unsigned short u16;
  float*    xn    = (float*)carve((size_t)NN * D * 4);
  float*    alpha = (float*)carve((size_t)E2 * 4);
  float*    exb   = (float*)carve((size_t)E2 * 4);
  unsigned* mxu   = (unsigned*)carve((size_t)NN * 4);
  float*    den   = (float*)carve((size_t)NN * 4);
  float*    h     = (float*)carve((size_t)NN * D * 4);
  u16*      hb    = (u16*)carve((size_t)NN * D * 2);
  u16*      qkvb  = (u16*)carve((size_t)NN * 768 * 2);
  u16*      attb  = (u16*)carve((size_t)NN * D * 2);
  float*    obuf  = (float*)carve((size_t)NN * D * 4);   // o, then ff2 out
  float*    x1    = (float*)carve((size_t)NN * D * 4);
  u16*      x1b   = (u16*)carve((size_t)NN * D * 2);
  u16*      x2b   = (u16*)carve((size_t)NN * D * 2);
  u16*      ffbb  = (u16*)carve((size_t)NN * FF * 2);
  u16*      fcbb  = (u16*)carve((size_t)NN * HD * 2);
  u16*      inWb  = (u16*)carve((size_t)768 * D * 2);
  u16*      opWb  = (u16*)carve((size_t)D * D * 2);
  u16*      f1Wb  = (u16*)carve((size_t)FF * D * 2);
  u16*      f2Wb  = (u16*)carve((size_t)D * FF * 2);
  u16*      fcWb  = (u16*)carve((size_t)HD * D * 2);
  u16*      oWb   = (u16*)carve((size_t)2 * HD * 2);

  hipMemsetAsync(mxu, 0, (size_t)NN * 4, stream);        // key(-inf) == 0
  hipMemsetAsync(den, 0, (size_t)NN * 4, stream);
  hipMemsetAsync(h,   0, (size_t)NN * D * 4, stream);

  // Weight bf16 conversions (stateless: redone every call)
  cvt_bf16<<<(768 * D + 255) / 256, 256, 0, stream>>>(inW, inWb, 768 * D);
  cvt_bf16<<<(D * D + 255) / 256, 256, 0, stream>>>(opW, opWb, D * D);
  cvt_bf16<<<(FF * D + 255) / 256, 256, 0, stream>>>(f1W, f1Wb, FF * D);
  cvt_bf16<<<(D * FF + 255) / 256, 256, 0, stream>>>(f2W, f2Wb, D * FF);
  cvt_bf16<<<(HD * D + 255) / 256, 256, 0, stream>>>(fcW, fcWb, HD * D);
  cvt_bf16<<<(2 * HD + 255) / 256, 256, 0, stream>>>(oW, oWb, 2 * HD);

  // AGNNConv
  l2norm_rows<<<NN, 256, 0, stream>>>(x, xn);
  edge_alpha<<<E2 / 8, 256, 0, stream>>>(xn, ei, beta, alpha, mxu, E, E2);
  edge_exp<<<(E2 + 255) / 256, 256, 0, stream>>>(alpha, ei, mxu, exb, den, E, E2);
  edge_scatter<<<E2 / 8, 256, 0, stream>>>(x, ei, exb, den, h, E, E2);
  cvt_bf16<<<(NN * D + 255) / 256, 256, 0, stream>>>(h, hb, NN * D);

  // Transformer encoder layer
  gemm_bf16<0, 1><<<dim3(NN / 64, 768 / 64), 128, 0, stream>>>(
      hb, inWb, inB, qkvb, NN, 768, D);
  attn_flash<<<dim3(NN / 64, 2), 128, 0, stream>>>(qkvb, attb);
  gemm_bf16<0, 0><<<dim3(NN / 64, D / 64), 128, 0, stream>>>(
      attb, opWb, opB, obuf, NN, D, D);
  ln_res<<<NN, 256, 0, stream>>>(h, obuf, g1, b1, x1, x1b);
  gemm_bf16<1, 1><<<dim3(NN / 64, FF / 64), 128, 0, stream>>>(
      x1b, f1Wb, f1B, ffbb, NN, FF, D);
  gemm_bf16<0, 0><<<dim3(NN / 64, D / 64), 128, 0, stream>>>(
      ffbb, f2Wb, f2B, obuf, NN, D, FF);
  ln_res<<<NN, 256, 0, stream>>>(x1, obuf, g2, b2, nullptr, x2b);

  // Head
  gemm_bf16<1, 1><<<dim3(NN / 64, HD / 64), 128, 0, stream>>>(
      x2b, fcWb, fcB, fcbb, NN, HD, D);
  gemm_bf16<0, 0><<<dim3(NN / 64, 1), 128, 0, stream>>>(
      fcbb, oWb, oB, out, NN, 2, HD);
}
